// EGATLayer_29274497090109
// MI455X (gfx1250) — matrix operand order repaired and checked by reading, not compiled, not run on hardware
//
#include <hip/hip_runtime.h>
#include <hip/hip_bf16.h>

#define NN 50000
#define EE 400000
#define IND 256
#define EF 64
#define OD 32
#define EO 64
#define NH 8
#define HD 256           // H * OUT_DIM
#define ALPHA_LR 0.2f
#define EPS_BN 1e-5f

typedef __attribute__((ext_vector_type(16))) _Float16 v16h;
typedef __attribute__((ext_vector_type(8)))  float    v8f;

__device__ __forceinline__ float lrelu(float x) { return x > 0.f ? x : ALPHA_LR * x; }

__device__ __forceinline__ v8f wmma16(v16h a, v16h b, v8f c) {
  // D = A(16x32 f16) * B(32x16 f16) + C(16x16 f32)
  return __builtin_amdgcn_wmma_f32_16x16x32_f16(false, a, false, b, (short)0, c,
                                                false, false);
}

// --- WMMA operand loaders -------------------------------------------------
// A tile from LDS (row-major f16, ldah = row stride in halves).
// ISA layout (16-bit A 16x32): lane l -> row m=l&15, k(i) = (i/8)*16 + (l>>4)*8 + i%8.
// Each 8-element group is 8 consecutive halves (16B aligned) -> ds_load_b128.
__device__ __forceinline__ v16h load_a_lds(const _Float16* lds, int ldah,
                                           int rbase, int k0, int lane) {
  const _Float16* p = lds + (rbase + (lane & 15)) * ldah + k0 + ((lane >> 4) << 3);
  v16h a;
#pragma unroll
  for (int j = 0; j < 8; ++j) a[j] = p[j];
#pragma unroll
  for (int j = 0; j < 8; ++j) a[8 + j] = p[16 + j];
  return a;
}

// B tile from pre-swizzled global f16: each lane's 16 halves are contiguous (32B).
__device__ __forceinline__ v16h load_bsw(const _Float16* bsw, int nt_count,
                                         int kt, int nt, int lane) {
  return *(const v16h*)(bsw + ((((size_t)kt * nt_count + nt) * 32 + lane) << 4));
}

// --- helpers --------------------------------------------------------------
__device__ __forceinline__ void atomicMaxF(float* addr, float v) {
  // monotonic int/uint mapping trick; addr pre-initialized to -3e38
  if (v >= 0.f) atomicMax((int*)addr, __float_as_int(v));
  else          atomicMin((unsigned int*)addr, __float_as_uint(v));
}

__global__ void k_fill(float* __restrict__ p, float v, int n) {
  int i = blockIdx.x * blockDim.x + threadIdx.x;
  if (i < n) p[i] = v;
}

// Convert fp32 weight [K][Ncol] -> f16 swizzled into WMMA B lane layout.
// ISA layout (16-bit B 32x16): lane l -> col n=l&15, k(i) = (i/8)*16 + (l>>4)*8 + i%8.
__global__ void k_swizzle(const float* __restrict__ w, _Float16* __restrict__ out,
                          int K, int Ncol) {
  int idx = blockIdx.x * blockDim.x + threadIdx.x;
  if (idx >= K * Ncol) return;
  int i    = idx & 15;
  int lane = (idx >> 4) & 31;
  int tl   = idx >> 9;
  int NT   = Ncol >> 4;
  int kt = tl / NT, nt = tl - kt * NT;
  int n = nt * 16 + (lane & 15);
  int k = kt * 32 + ((i >> 3) << 4) + ((lane >> 4) << 3) + (i & 7);
  out[idx] = (_Float16)w[k * Ncol + n];
}

// --- 1. node projection: ft = node_inputs(N,256) @ fc_w(256,256) ----------
__global__ __launch_bounds__(256) void k_gemm_node(const float* __restrict__ A,
                                                   const _Float16* __restrict__ Bsw,
                                                   float* __restrict__ C) {
  __shared__ _Float16 ldsA[32 * 256];            // 16 KB
  int row0 = blockIdx.x * 32;
  int t = threadIdx.x;
  for (int idx = t; idx < 32 * 256; idx += 256) {
    int r = idx >> 8, c = idx & 255;
    int row = row0 + r;
    ldsA[idx] = (_Float16)((row < NN) ? A[(size_t)row * 256 + c] : 0.f);
  }
  __syncthreads();
  int wave = t >> 5, lane = t & 31;
  int rt = wave >> 2, ct = wave & 3;             // 8 waves -> 2 row x 4 col tiles
  int nt = blockIdx.y * 4 + ct;                  // 16 col-tiles total
  v8f acc = {};
#pragma unroll
  for (int kt = 0; kt < 8; ++kt) {
    v16h a = load_a_lds(ldsA, 256, rt * 16, kt * 32, lane);
    v16h b = load_bsw(Bsw, 16, kt, nt, lane);
    acc = wmma16(a, b, acc);
  }
  int col = nt * 16 + (lane & 15);
  int rr  = rt * 16 + ((lane >> 4) << 3);
#pragma unroll
  for (int r = 0; r < 8; ++r) {
    int row = row0 + rr + r;
    if (row < NN) C[(size_t)row * 256 + col] = acc[r];
  }
}

// --- 2. per-head attention scalars ---------------------------------------
__global__ void k_attn(const float* __restrict__ ft, const float* __restrict__ al,
                       const float* __restrict__ ar, float* __restrict__ a1,
                       float* __restrict__ a2) {
  int idx = blockIdx.x * blockDim.x + threadIdx.x;  // n*8 + h
  if (idx >= NN * NH) return;
  int h = idx & 7;
  const float* f = ft + (size_t)idx * OD;
  float s1 = 0.f, s2 = 0.f;
#pragma unroll
  for (int d = 0; d < OD; ++d) {
    float v = f[d];
    s1 += v * al[h * OD + d];
    s2 += v * ar[h * OD + d];
  }
  a1[idx] = s1;
  a2[idx] = s2;
}

// --- 3. edge-feature attention MLP (skinny: VALU) -------------------------
__global__ __launch_bounds__(256) void k_edge_mlp(const float* __restrict__ ein,
                                                  const float* __restrict__ w1,
                                                  const float* __restrict__ b1,
                                                  const float* __restrict__ w2,
                                                  const float* __restrict__ b2,
                                                  float* __restrict__ eattn) {
  __shared__ float sw1[EF * OD], sb1[OD], sw2[OD * NH], sb2[NH];
  int t = threadIdx.x;
  for (int i = t; i < EF * OD; i += 256) sw1[i] = w1[i];
  for (int i = t; i < OD * NH; i += 256) sw2[i] = w2[i];
  if (t < OD) sb1[t] = b1[t];
  if (t < NH) sb2[t] = b2[t];
  __syncthreads();
  int e = blockIdx.x * 256 + t;
  if (e >= EE) return;
  float x[EF];
#pragma unroll
  for (int k = 0; k < EF; ++k) x[k] = ein[(size_t)e * EF + k];
  float hid[OD];
#pragma unroll
  for (int j = 0; j < OD; ++j) {
    float s = sb1[j];
#pragma unroll
    for (int k = 0; k < EF; ++k) s += x[k] * sw1[k * OD + j];
    hid[j] = lrelu(s);
  }
#pragma unroll
  for (int j = 0; j < NH; ++j) {
    float s = sb2[j];
#pragma unroll
    for (int k = 0; k < OD; ++k) s += hid[k] * sw2[k * NH + j];
    eattn[(size_t)e * NH + j] = s;
  }
}

// --- 4. logits + segment max ----------------------------------------------
__global__ void k_logits(const float* __restrict__ a1, const float* __restrict__ a2,
                         const int* __restrict__ src, const int* __restrict__ dst,
                         float* __restrict__ apre, float* __restrict__ amax) {
  int idx = blockIdx.x * blockDim.x + threadIdx.x;
  if (idx >= EE * NH) return;
  int e = idx >> 3, h = idx & 7;
  int s = src[e], d = dst[e];
  float v = lrelu(a1[s * NH + h] + a2[d * NH + h] + apre[idx]);  // TEMP = 1
  apre[idx] = v;
  atomicMaxF(&amax[d * NH + h], v);
}

// --- 5. exp + segment sum -------------------------------------------------
__global__ void k_exp(const int* __restrict__ dst, float* __restrict__ a,
                      const float* __restrict__ amax, float* __restrict__ z) {
  int idx = blockIdx.x * blockDim.x + threadIdx.x;
  if (idx >= EE * NH) return;
  int e = idx >> 3, h = idx & 7;
  int d = dst[e];
  float v = expf(a[idx] - amax[d * NH + h]);
  a[idx] = v;
  atomicAdd(&z[d * NH + h], v);
}

// --- 6. weighted message aggregation --------------------------------------
__global__ void k_agg(const float* __restrict__ ft, const float* __restrict__ a,
                      const int* __restrict__ src, const int* __restrict__ dst,
                      float* __restrict__ agg) {
  size_t idx = (size_t)blockIdx.x * 256 + threadIdx.x;  // e*64 + q (float4 lanes)
  if (idx >= (size_t)EE * 64) return;
  int e = (int)(idx >> 6), q = (int)(idx & 63);
  int s = src[e], d = dst[e];
  float av = a[(size_t)e * NH + (q >> 3)];
  float4 f = *(const float4*)(ft + (size_t)s * HD + q * 4);
  float* o = agg + (size_t)d * HD + q * 4;
  atomicAdd(o + 0, f.x * av);
  atomicAdd(o + 1, f.y * av);
  atomicAdd(o + 2, f.z * av);
  atomicAdd(o + 3, f.w * av);
}

// --- 7. normalize + residual + lrelu + BN partial sums --------------------
__global__ __launch_bounds__(256) void k_node_post(const float* __restrict__ agg,
                                                   const float* __restrict__ z,
                                                   const float* __restrict__ nin,
                                                   float* __restrict__ outn,
                                                   float* __restrict__ sums) {
  int c = threadIdx.x;          // channel 0..255, h = c>>5
  int n0 = blockIdx.x * 64;
  int h = c >> 5;
  float s = 0.f, s2 = 0.f;
  for (int i = 0; i < 64; ++i) {
    int n = n0 + i;
    if (n >= NN) break;
    float zz = z[n * NH + h];
    if (zz == 0.f) zz = 1.f;
    float x = agg[(size_t)n * HD + c] / zz + nin[(size_t)n * HD + c];
    x = lrelu(x);
    outn[(size_t)n * HD + c] = x;
    s += x;
    s2 += x * x;
  }
  atomicAdd(&sums[c], s);
  atomicAdd(&sums[HD + c], s2);
}

// --- 8. BN finalize / apply -----------------------------------------------
__global__ void k_bn_fin(const float* __restrict__ sums, float* __restrict__ musig,
                         int C, float cnt) {
  int c = blockIdx.x * blockDim.x + threadIdx.x;
  if (c >= C) return;
  float mu = sums[c] / cnt;
  float var = sums[C + c] / cnt - mu * mu;
  musig[c] = mu;
  musig[C + c] = rsqrtf(var + EPS_BN);
}

__global__ void k_bn_apply(float* __restrict__ x, const float* __restrict__ musig,
                           const float* __restrict__ g, const float* __restrict__ b,
                           int C, size_t total) {
  size_t i = (size_t)blockIdx.x * blockDim.x + threadIdx.x;
  if (i >= total) return;
  int c = (int)(i & (size_t)(C - 1));                 // C is a power of two
  x[i] = (x[i] - musig[c]) * musig[C + c] * g[c] + b[c];
}

// --- 9. fused edge update: 3 chained WMMA GEMMs per 32-edge tile ----------
__global__ __launch_bounds__(256) void k_edge_update(
    const float* __restrict__ nodef, const float* __restrict__ ein,
    const _Float16* __restrict__ wnode_sw, const _Float16* __restrict__ wedge_sw,
    const _Float16* __restrict__ wfinal_sw, const int* __restrict__ src,
    const int* __restrict__ dst, float* __restrict__ oute,
    float* __restrict__ esums) {
  __shared__ _Float16 ldsA[32 * 512];   // 32 KB gathered concat rows (f16)
  __shared__ _Float16 ldsC[32 * 128];   // 8 KB  concat(node_f | edge_f)
  __shared__ float    ldsO[32 * 64];    // 8 KB  outputs for BN partials
  __shared__ int se[32], de[32];
  int t = threadIdx.x;
  int e0 = blockIdx.x * 32;
  if (t < 32) { se[t] = src[e0 + t]; de[t] = dst[e0 + t]; }
  __syncthreads();
  // stage A1 = [node_out[src] | node_out[dst]]  (32 x 512) as f16
  for (int idx = t; idx < 32 * 512; idx += 256) {
    int r = idx >> 9, c = idx & 511;
    int row = (c < 256) ? se[r] : de[r];
    ldsA[idx] = (_Float16)nodef[(size_t)row * 256 + (c & 255)];
  }
  __syncthreads();
  int wave = t >> 5, lane = t & 31;
  int rt = wave >> 2, ct = wave & 3;    // 2 row-tiles x 4 col-tiles
  // phase 1: node_f = lrelu(A1 @ eu_node_w)   K=512
  {
    v8f acc = {};
#pragma unroll
    for (int kt = 0; kt < 16; ++kt) {
      v16h a = load_a_lds(ldsA, 512, rt * 16, kt * 32, lane);
      v16h b = load_bsw(wnode_sw, 4, kt, ct, lane);
      acc = wmma16(a, b, acc);
    }
    int col = ct * 16 + (lane & 15);
    int rr  = rt * 16 + ((lane >> 4) << 3);
#pragma unroll
    for (int r = 0; r < 8; ++r) ldsC[(rr + r) * 128 + col] = (_Float16)lrelu(acc[r]);
  }
  __syncthreads();
  // phase 2: edge_f = lrelu(edge_in @ eu_edge_w)   K=64 (reuse ldsA head)
  for (int idx = t; idx < 32 * 64; idx += 256) {
    int r = idx >> 6, c = idx & 63;
    ldsA[idx] = (_Float16)ein[(size_t)(e0 + r) * 64 + c];
  }
  __syncthreads();
  {
    v8f acc = {};
#pragma unroll
    for (int kt = 0; kt < 2; ++kt) {
      v16h a = load_a_lds(ldsA, 64, rt * 16, kt * 32, lane);
      v16h b = load_bsw(wedge_sw, 4, kt, ct, lane);
      acc = wmma16(a, b, acc);
    }
    int col = 64 + ct * 16 + (lane & 15);
    int rr  = rt * 16 + ((lane >> 4) << 3);
#pragma unroll
    for (int r = 0; r < 8; ++r) ldsC[(rr + r) * 128 + col] = (_Float16)lrelu(acc[r]);
  }
  __syncthreads();
  // phase 3: edge_pre = concat @ eu_final_w + edge_in   K=128
  {
    v8f acc = {};
#pragma unroll
    for (int kt = 0; kt < 4; ++kt) {
      v16h a = load_a_lds(ldsC, 128, rt * 16, kt * 32, lane);
      v16h b = load_bsw(wfinal_sw, 4, kt, ct, lane);
      acc = wmma16(a, b, acc);
    }
    int col = ct * 16 + (lane & 15);
    int rr  = rt * 16 + ((lane >> 4) << 3);
#pragma unroll
    for (int r = 0; r < 8; ++r) {
      int row = rr + r;
      float v = acc[r] + ein[(size_t)(e0 + row) * 64 + col];
      oute[(size_t)(e0 + row) * 64 + col] = v;
      ldsO[row * 64 + col] = v;
    }
  }
  __syncthreads();
  if (t < 64) {
    float s = 0.f, s2 = 0.f;
#pragma unroll
    for (int r = 0; r < 32; ++r) { float v = ldsO[r * 64 + t]; s += v; s2 += v * v; }
    atomicAdd(&esums[t], s);
    atomicAdd(&esums[64 + t], s2);
  }
}

// --------------------------------------------------------------------------
extern "C" void kernel_launch(void* const* d_in, const int* in_sizes, int n_in,
                              void* d_out, int out_size, void* d_ws, size_t ws_size,
                              hipStream_t stream) {
  (void)in_sizes; (void)n_in; (void)out_size; (void)ws_size;
  const float* node_in   = (const float*)d_in[0];
  const float* edge_in   = (const float*)d_in[1];
  const float* fc_w      = (const float*)d_in[2];
  const float* attn_l    = (const float*)d_in[3];
  const float* attn_r    = (const float*)d_in[4];
  const float* ae_w1     = (const float*)d_in[5];
  const float* ae_b1     = (const float*)d_in[6];
  const float* ae_w2     = (const float*)d_in[7];
  const float* ae_b2     = (const float*)d_in[8];
  const float* bn_n_g    = (const float*)d_in[9];
  const float* bn_n_b    = (const float*)d_in[10];
  const float* eu_node_w = (const float*)d_in[11];
  const float* eu_edge_w = (const float*)d_in[12];
  const float* eu_final_w= (const float*)d_in[13];
  const float* bn_e_g    = (const float*)d_in[14];
  const float* bn_e_b    = (const float*)d_in[15];
  const int*   src       = (const int*)d_in[16];
  const int*   dst       = (const int*)d_in[17];

  float* out_node = (float*)d_out;
  float* out_edge = out_node + (size_t)NN * HD;

  float* ws = (float*)d_ws;
  float* ft    = ws;  ws += (size_t)NN * HD;
  float* a1    = ws;  ws += NN * NH;
  float* a2    = ws;  ws += NN * NH;
  float* eattn = ws;  ws += (size_t)EE * NH;   // reused for logits, then exp(a)
  float* amax  = ws;  ws += NN * NH;
  float* zsum  = ws;  ws += NN * NH;
  float* agg   = ws;  ws += (size_t)NN * HD;
  float* sums_n  = ws; ws += 512;
  float* musig_n = ws; ws += 512;
  float* sums_e  = ws; ws += 128;
  float* musig_e = ws; ws += 128;
  _Float16* fcw_sw    = (_Float16*)ws;
  _Float16* wnode_sw  = fcw_sw + 256 * 256;
  _Float16* wedge_sw  = wnode_sw + 512 * 64;
  _Float16* wfinal_sw = wedge_sw + 64 * 64;

  // zero z, agg, BN stat buffers (contiguous) each call; fill amax with -inf
  size_t zero_floats = (size_t)NN * NH + (size_t)NN * HD + 512 + 512 + 128 + 128;
  (void)hipMemsetAsync(zsum, 0, zero_floats * sizeof(float), stream);
  k_fill<<<(NN * NH + 255) / 256, 256, 0, stream>>>(amax, -3.0e38f, NN * NH);

  // pre-convert + pre-swizzle weights into WMMA B lane layout (f16)
  k_swizzle<<<(256 * 256 + 255) / 256, 256, 0, stream>>>(fc_w, fcw_sw, 256, 256);
  k_swizzle<<<(512 * 64 + 255) / 256, 256, 0, stream>>>(eu_node_w, wnode_sw, 512, 64);
  k_swizzle<<<(64 * 64 + 255) / 256, 256, 0, stream>>>(eu_edge_w, wedge_sw, 64, 64);
  k_swizzle<<<(128 * 64 + 255) / 256, 256, 0, stream>>>(eu_final_w, wfinal_sw, 128, 64);

  dim3 g_ft((NN + 31) / 32, 4);
  k_gemm_node<<<g_ft, 256, 0, stream>>>(node_in, fcw_sw, ft);
  k_attn<<<(NN * NH + 255) / 256, 256, 0, stream>>>(ft, attn_l, attn_r, a1, a2);
  k_edge_mlp<<<(EE + 255) / 256, 256, 0, stream>>>(edge_in, ae_w1, ae_b1, ae_w2,
                                                   ae_b2, eattn);
  k_logits<<<(EE * NH + 255) / 256, 256, 0, stream>>>(a1, a2, src, dst, eattn, amax);
  k_exp<<<(EE * NH + 255) / 256, 256, 0, stream>>>(dst, eattn, amax, zsum);
  k_agg<<<(unsigned)((size_t)EE * 64 / 256), 256, 0, stream>>>(ft, eattn, src, dst, agg);
  k_node_post<<<(NN + 63) / 64, 256, 0, stream>>>(agg, zsum, node_in, out_node, sums_n);
  k_bn_fin<<<1, 256, 0, stream>>>(sums_n, musig_n, 256, (float)NN);
  k_bn_apply<<<(unsigned)(((size_t)NN * HD + 255) / 256), 256, 0, stream>>>(
      out_node, musig_n, bn_n_g, bn_n_b, 256, (size_t)NN * HD);
  k_edge_update<<<EE / 32, 256, 0, stream>>>(out_node, edge_in, wnode_sw, wedge_sw,
                                             wfinal_sw, src, dst, out_edge, sums_e);
  k_bn_fin<<<1, 64, 0, stream>>>(sums_e, musig_e, 64, (float)EE);
  k_bn_apply<<<(unsigned)(((size_t)EE * EO + 255) / 256), 256, 0, stream>>>(
      out_edge, musig_e, bn_e_g, bn_e_b, 64, (size_t)EE * EO);
}
